// Cortex_30794915512832
// MI455X (gfx1250) — matrix operand order repaired and checked by reading, not compiled, not run on hardware
//
#include <hip/hip_runtime.h>
#include <hip/hip_bf16.h>

// ---------------- problem constants (from reference) ----------------
#define B_DIM 1024
#define C_DIM 64
#define D_DIM 256
#define N_DIM 512
#define K1    (2 * D_DIM)     // fused K for drive GEMM (x_in || x_ctx)
#define FEEDBACK 0.3f

// ---------------- WMMA types (CDNA5 gfx1250, wave32) ----------------
typedef __attribute__((ext_vector_type(16))) __bf16 v16bf;
typedef __attribute__((ext_vector_type(8)))  float  v8f;
typedef int v4i_vs __attribute__((vector_size(16)));   // matches async builtin params

union FragU { uint4 u[2]; v16bf v; };
union PackU { __bf16 b[4]; uint2 u; };

// LDS row stride in bf16 elements; multiple of 8 so every 8-element slice is
// 16-byte aligned for ds_load_b128 / async b128 stores.
#define LSTR 40

#define AS1 __attribute__((address_space(1)))
#define AS3 __attribute__((address_space(3)))

#if __has_builtin(__builtin_amdgcn_global_load_async_to_lds_b128) && \
    __has_builtin(__builtin_amdgcn_s_wait_asynccnt)
#define ASYNC_OK 1
#else
#define ASYNC_OK 0
#endif

// 16-byte global -> LDS copy. On gfx1250 this is GLOBAL_LOAD_ASYNC_TO_LDS_B128
// (tracked by ASYNCcnt, no VGPR round trip); fallback is a plain VGPR copy.
__device__ __forceinline__ void async_cp16(const void* g, void* l) {
#if ASYNC_OK
    __builtin_amdgcn_global_load_async_to_lds_b128(
        (AS1 v4i_vs*)g, (AS3 v4i_vs*)l, 0, 0);
#else
    *(uint4*)l = *(const uint4*)g;
#endif
}
__device__ __forceinline__ void async_join() {
#if ASYNC_OK
    __builtin_amdgcn_s_wait_asynccnt(0);
#endif
}

// A fragment, 16x32 bf16 (ISA 7.12.2):
//   lanes 0-15 : row m, K = 0..7 & 16..23 ; lanes 16-31: row m, K = 8..15 & 24..31
__device__ __forceinline__ v16bf load_frag_A(const __bf16* As, int m, int khalf) {
    FragU f;
    f.u[0] = *reinterpret_cast<const uint4*>(As + m * LSTR + khalf * 8);
    f.u[1] = *reinterpret_cast<const uint4*>(As + m * LSTR + 16 + khalf * 8);
    return f.v;
}
// B fragment, 32x16 bf16, LDS tile stored [n][k] (K contiguous):
//   lanes 0-15 : col n, K = 0..7 & 16..23 ; lanes 16-31 : col n, K = 8..15 & 24..31
__device__ __forceinline__ v16bf load_frag_B(const __bf16* Bs, int n, int khalf) {
    FragU f;
    f.u[0] = *reinterpret_cast<const uint4*>(Bs + n * LSTR + khalf * 8);
    f.u[1] = *reinterpret_cast<const uint4*>(Bs + n * LSTR + 16 + khalf * 8);
    return f.v;
}

// =====================================================================
// Prep 1: tiled transpose + fp32->bf16.  out[c][j][i] = scale * in[c][i][j]
// in: [C][RI][RJ] (j contiguous) ; out: stride `ostr` along j-rows, i contiguous.
// grid = (RI/32, RJ/32, C), block = (32, 8)
// =====================================================================
__global__ __launch_bounds__(256)
void cortex_transpose_bf16(const float* __restrict__ in, __bf16* __restrict__ out,
                           int RI, int RJ, int ostr, float scale) {
    __shared__ __bf16 tile[32][33];
    const int i0 = blockIdx.x * 32;
    const int j0 = blockIdx.y * 32;
    const int c  = blockIdx.z;
    const float*  src = in  + (size_t)c * RI * RJ;
    __bf16*       dst = out + (size_t)c * RJ * ostr;
    #pragma unroll
    for (int r = threadIdx.y; r < 32; r += 8)      // r = i
        tile[r][threadIdx.x] =
            (__bf16)(src[(size_t)(i0 + r) * RJ + j0 + threadIdx.x] * scale);
    __syncthreads();
    #pragma unroll
    for (int r = threadIdx.y; r < 32; r += 8)      // r = j
        dst[(size_t)(j0 + r) * ostr + i0 + threadIdx.x] = tile[threadIdx.x][r];
}

// Prep 2: xb[b][k] = bf16( k<256 ? x_in[b][k] : x_ctx[b][k-256] )
__global__ __launch_bounds__(256)
void cortex_pack_x(const float* __restrict__ x_in, const float* __restrict__ x_ctx,
                   __bf16* __restrict__ xb) {
    const int idx = blockIdx.x * 256 + threadIdx.x;     // B*K1 total
    const int b = idx >> 9, kq = idx & (K1 - 1);
    const float v = (kq < D_DIM) ? x_in[(size_t)b * D_DIM + kq]
                                 : x_ctx[(size_t)b * D_DIM + kq - D_DIM];
    xb[idx] = (__bf16)v;
}

// =====================================================================
// Kernel 1: drive[b,c,n] = xb · Wt1[c] + bias   (bf16 WMMA, K = 512)
// xb:  [B][512] bf16 (x_in || x_ctx, 0.3 folded into Wt1)
// Wt1: [C][N][512] bf16 (K contiguous)
// grid = (N/64, B/64, C), block = 128 (4 waves); wave w owns rows w*16..+15.
// Writes drive into activation region of d_out (kwta overwrites in place).
// =====================================================================
__global__ __launch_bounds__(128)
void cortex_gemm_drive(const __bf16* __restrict__ xb,
                       const __bf16* __restrict__ Wt1,
                       const float* __restrict__ bias,
                       float* __restrict__ drive_out) {
    __shared__ __bf16 As[64 * LSTR];
    __shared__ __bf16 Bs[64 * LSTR];

    const int c    = blockIdx.z;
    const int b0   = blockIdx.y * 64;
    const int n0   = blockIdx.x * 64;
    const int tid  = threadIdx.x;
    const int lane = tid & 31;
    const int wave = tid >> 5;

    const __bf16* wc = Wt1 + (size_t)c * N_DIM * K1;

    v8f acc[4];
    #pragma unroll
    for (int t = 0; t < 4; ++t)
        #pragma unroll
        for (int e = 0; e < 8; ++e) acc[t][e] = 0.0f;

    for (int kk = 0; kk < K1; kk += 32) {
        // stage A (64x32 bf16) and B (64x32 bf16) tiles: 2 x b128 async each
        #pragma unroll
        for (int h = 0; h < 2; ++h) {
            const int q = tid + 128 * h;            // 16B chunk id, 256 total
            const int r = q >> 2, c8 = q & 3;       // row, 8-elem column group
            async_cp16(xb + (size_t)(b0 + r) * K1 + kk + c8 * 8,
                       &As[r * LSTR + c8 * 8]);
            async_cp16(wc + (size_t)(n0 + r) * K1 + kk + c8 * 8,
                       &Bs[r * LSTR + c8 * 8]);
        }
        async_join();
        __syncthreads();

        const int kh = lane >> 4;
        const v16bf a = load_frag_A(As, wave * 16 + (lane & 15), kh);
        #pragma unroll
        for (int t = 0; t < 4; ++t) {
            const v16bf b = load_frag_B(Bs, t * 16 + (lane & 15), kh);
            acc[t] = __builtin_amdgcn_wmma_f32_16x16x32_bf16(
                false, a, false, b, (short)0, acc[t], false, false);
        }
        __syncthreads();
    }

    const int kh = lane >> 4;
    #pragma unroll
    for (int t = 0; t < 4; ++t) {
        const int n = n0 + t * 16 + (lane & 15);
        const float bv = bias[c * N_DIM + n];
        #pragma unroll
        for (int r = 0; r < 8; ++r) {
            const int m = b0 + wave * 16 + r + 8 * kh;
            drive_out[((size_t)m * C_DIM + c) * N_DIM + n] = acc[t][r] + bv;
        }
    }
}

// =====================================================================
// Kernel 2: k-winners-take-all + homeostatic boost + normalize, in place.
// One wave32 per (b,c) row of N=512 (16 values/lane); exact k-th threshold
// via k wave-max extractions (__shfl_xor width 32, __ballot to delete one).
// =====================================================================
__global__ __launch_bounds__(256)
void cortex_kwta(float* __restrict__ act,
                 const float* __restrict__ avg_activity,
                 const int* __restrict__ kptr) {
    const int tid  = threadIdx.x;
    const int lane = tid & 31;
    const int wave = tid >> 5;
    const int rid  = blockIdx.x * 8 + wave;      // rid = b*C + c
    const int c    = rid & (C_DIM - 1);
    const int k    = *kptr;

    float* row = act + (size_t)rid * N_DIM;

    float drive[16], boosted[16], work[16];
    #pragma unroll
    for (int j = 0; j < 16; ++j) {
        const int n = lane + 32 * j;
        const float d = row[n];
        const float a = avg_activity[c * N_DIM + n];
        const float bst = log1pf(0.05f / (a + 1e-6f));   // * (0.01*100) == 1.0
        drive[j]   = d;
        boosted[j] = d + bst;
        work[j]    = boosted[j];
    }

    float thr = 0.0f;
    for (int it = 0; it < k; ++it) {
        float m = -__builtin_inff();
        #pragma unroll
        for (int j = 0; j < 16; ++j) m = fmaxf(m, work[j]);
        #pragma unroll
        for (int off = 16; off >= 1; off >>= 1)
            m = fmaxf(m, __shfl_xor(m, off, 32));
        thr = m;
        int jloc = -1;
        #pragma unroll
        for (int j = 0; j < 16; ++j)
            if (jloc < 0 && work[j] == m) jloc = j;
        const unsigned long long bal = __ballot(jloc >= 0);
        const int src = __ffsll((unsigned long long)bal) - 1;
        if (lane == src && jloc >= 0) work[jloc] = -__builtin_inff();
    }

    float a16[16];
    float sum = 0.0f;
    #pragma unroll
    for (int j = 0; j < 16; ++j) {
        const float v = (boosted[j] >= thr) ? fmaxf(drive[j], 0.0f) : 0.0f;
        a16[j] = v;
        sum += v;
    }
    #pragma unroll
    for (int off = 16; off >= 1; off >>= 1)
        sum += __shfl_xor(sum, off, 32);
    const float scale = (float)k / (sum + 1e-8f);
    #pragma unroll
    for (int j = 0; j < 16; ++j) row[lane + 32 * j] = a16[j] * scale;
}

// =====================================================================
// Kernel 3: predictions[b,c,d] = act · Wt2[c] ; errors = x_in - pred
// act: fp32 [B][C][N] (converted to bf16 tile-by-tile with float4 loads)
// Wt2: [C][D][512] bf16 (K = n contiguous), async-staged.
// grid = (D/64, B/64, C), block = 128 (4 waves). K = N = 512.
// =====================================================================
__global__ __launch_bounds__(128)
void cortex_gemm_pred(const float* __restrict__ act,
                      const __bf16* __restrict__ Wt2,
                      const float* __restrict__ x_in,
                      float* __restrict__ pred,
                      float* __restrict__ err) {
    __shared__ __bf16 As[64 * LSTR];
    __shared__ __bf16 Bs[64 * LSTR];

    const int c    = blockIdx.z;
    const int b0   = blockIdx.y * 64;
    const int d0g  = blockIdx.x * 64;
    const int tid  = threadIdx.x;
    const int lane = tid & 31;
    const int wave = tid >> 5;

    const __bf16* wc = Wt2 + (size_t)c * D_DIM * N_DIM;

    v8f acc[4];
    #pragma unroll
    for (int t = 0; t < 4; ++t)
        #pragma unroll
        for (int e = 0; e < 8; ++e) acc[t][e] = 0.0f;

    for (int kk = 0; kk < N_DIM; kk += 32) {
        // B tile (64 d x 32 k bf16): async b128 x2 per thread
        #pragma unroll
        for (int h = 0; h < 2; ++h) {
            const int q = tid + 128 * h;
            const int r = q >> 2, c8 = q & 3;
            async_cp16(wc + (size_t)(d0g + r) * N_DIM + kk + c8 * 8,
                       &Bs[r * LSTR + c8 * 8]);
        }
        // A tile (64 b x 32 k): float4 activation load -> packed 4xbf16 store
        {
            const int t4 = tid & 7;                 // float4 column (8 per row)
            const int r0 = tid >> 3;                // 0..15
            #pragma unroll
            for (int i = 0; i < 4; ++i) {
                const int r = r0 + 16 * i;
                const float4 v = *reinterpret_cast<const float4*>(
                    &act[((size_t)(b0 + r) * C_DIM + c) * N_DIM + kk + t4 * 4]);
                PackU p;
                p.b[0] = (__bf16)v.x; p.b[1] = (__bf16)v.y;
                p.b[2] = (__bf16)v.z; p.b[3] = (__bf16)v.w;
                *reinterpret_cast<uint2*>(&As[r * LSTR + t4 * 4]) = p.u;
            }
        }
        async_join();
        __syncthreads();

        const int kh = lane >> 4;
        const v16bf a = load_frag_A(As, wave * 16 + (lane & 15), kh);
        #pragma unroll
        for (int t = 0; t < 4; ++t) {
            const v16bf b = load_frag_B(Bs, t * 16 + (lane & 15), kh);
            acc[t] = __builtin_amdgcn_wmma_f32_16x16x32_bf16(
                false, a, false, b, (short)0, acc[t], false, false);
        }
        __syncthreads();
    }

    const int kh = lane >> 4;
    #pragma unroll
    for (int t = 0; t < 4; ++t) {
        const int d = d0g + t * 16 + (lane & 15);
        #pragma unroll
        for (int r = 0; r < 8; ++r) {
            const int m = b0 + wave * 16 + r + 8 * kh;
            const float p = acc[t][r];
            const size_t o = ((size_t)m * C_DIM + c) * D_DIM + d;
            pred[o] = p;
            err[o]  = x_in[(size_t)m * D_DIM + d] - p;
        }
    }
}

// =====================================================================
extern "C" void kernel_launch(void* const* d_in, const int* in_sizes, int n_in,
                              void* d_out, int out_size, void* d_ws, size_t ws_size,
                              hipStream_t stream) {
    (void)in_sizes; (void)n_in; (void)out_size; (void)ws_size;

    const float* x_in  = (const float*)d_in[0];
    const float* x_ctx = (const float*)d_in[1];
    const float* Wff   = (const float*)d_in[2];
    const float* Wctx  = (const float*)d_in[3];
    const float* Wpred = (const float*)d_in[4];
    const float* bias  = (const float*)d_in[5];
    const float* avg   = (const float*)d_in[6];
    const int*   kptr  = (const int*)d_in[7];

    float* act  = (float*)d_out;                                   // [B,C,N]
    float* pred = act  + (size_t)B_DIM * C_DIM * N_DIM;            // [B,C,D]
    float* err  = pred + (size_t)B_DIM * C_DIM * D_DIM;            // [B,C,D]

    // workspace: bf16, K-contiguous operands (~49 MB)
    __bf16* Wt1 = (__bf16*)d_ws;                                   // [C][N][512]
    __bf16* Wt2 = Wt1 + (size_t)C_DIM * N_DIM * K1;                // [C][D][512]
    __bf16* xb  = Wt2 + (size_t)C_DIM * D_DIM * N_DIM;             // [B][512]

    // ---- prep: convert + transpose into workspace ----
    dim3 tb(32, 8);
    dim3 tg1(D_DIM / 32, N_DIM / 32, C_DIM);     // (8,16,64)
    cortex_transpose_bf16<<<tg1, tb, 0, stream>>>(Wff,  Wt1,          D_DIM, N_DIM, K1, 1.0f);
    cortex_transpose_bf16<<<tg1, tb, 0, stream>>>(Wctx, Wt1 + D_DIM,  D_DIM, N_DIM, K1, FEEDBACK);
    dim3 tg2(N_DIM / 32, D_DIM / 32, C_DIM);     // (16,8,64)
    cortex_transpose_bf16<<<tg2, tb, 0, stream>>>(Wpred, Wt2,         N_DIM, D_DIM, N_DIM, 1.0f);
    cortex_pack_x<<<(B_DIM * K1) / 256, 256, 0, stream>>>(x_in, x_ctx, xb);

    // ---- main pipeline ----
    dim3 g1(N_DIM / 64, B_DIM / 64, C_DIM);      // (8,16,64)
    cortex_gemm_drive<<<g1, 128, 0, stream>>>(xb, Wt1, bias, act);

    cortex_kwta<<<(B_DIM * C_DIM) / 8, 256, 0, stream>>>(act, avg, kptr);

    dim3 g2(D_DIM / 64, B_DIM / 64, C_DIM);      // (4,16,64)
    cortex_gemm_pred<<<g2, 128, 0, stream>>>(act, Wt2, x_in, pred, err);
}
